// AdaConv_9139690406001
// MI455X (gfx1250) — matrix operand order repaired and checked by reading, not compiled, not run on hardware
//
#include <hip/hip_runtime.h>
#include <hip/hip_bf16.h>

// ---------------- problem constants ----------------
#define NB   8      // batch
#define CHN  512    // channels
#define HH   128
#define WW   128
#define SD   512    // style dim
#define CIG  8      // channels per group (in and out)
#define NGRP 64     // groups per sample
#define PIX  (HH*WW)
#define KPAD 96     // 72 real K (8ci x 9taps) padded to 3 WMMA slices

#define XS_MAIN (CIG * 10 * 132)   // 10560 bf16: 8 ch x 10 padded rows x (130 cols, stride 132)
#define XS_ZPAD 1056               // zero region absorbing off + wave*132 + tx for padded K
#define XS_TOT  (XS_MAIN + XS_ZPAD)

typedef __attribute__((ext_vector_type(16))) __bf16 v16bf;
typedef __attribute__((ext_vector_type(8)))  float  v8f;

// workspace layout (float units unless noted)
#define WS_SPOOL 0                 // 4096 f32 : s_pool[n][s]
#define WS_MEAN  4096              // 4096 f32 : per (n,c) mean
#define WS_RSTD  8192              // 4096 f32 : per (n,c) 1/std
#define WS_BIAS  12288             // 4096 f32 : pw_bias[n][c]
#define WS_KEFF_BYTES 65536        // then 512 groups * 16 * 96 bf16

__device__ __forceinline__ unsigned short f2bf(float f) {
    union { float f; unsigned int u; } x; x.f = f;
    unsigned int u = x.u;
    unsigned int r = (u + 0x7FFFu + ((u >> 16) & 1u)) >> 16;  // RNE
    return (unsigned short)r;
}

// ---------- 1) avg-pool style 4x4 -> s_pool ----------
__global__ void spool_kernel(const float* __restrict__ style, float* __restrict__ spool) {
    int g = blockIdx.x * 256 + threadIdx.x;           // g = n*512 + s
    if (g >= NB * SD) return;
    const float* p = style + (size_t)g * 16;
    float s = 0.f;
    #pragma unroll
    for (int i = 0; i < 16; ++i) s += p[i];
    spool[g] = s * (1.f / 16.f);
}

// ---------- 2) instance-norm stats (unbiased var, eps 1e-5) ----------
__global__ void stats_kernel(const float* __restrict__ pred,
                             float* __restrict__ meanw, float* __restrict__ rstdw) {
    int nc = blockIdx.x;
    const float* p = pred + (size_t)nc * PIX;
    float s = 0.f, s2 = 0.f;
    for (int i = threadIdx.x; i < PIX; i += 256) { float v = p[i]; s += v; s2 += v * v; }
    __shared__ float rs[256], rq[256];
    rs[threadIdx.x] = s; rq[threadIdx.x] = s2; __syncthreads();
    for (int o = 128; o > 0; o >>= 1) {
        if (threadIdx.x < o) { rs[threadIdx.x] += rs[threadIdx.x + o]; rq[threadIdx.x] += rq[threadIdx.x + o]; }
        __syncthreads();
    }
    if (threadIdx.x == 0) {
        float sum = rs[0], ssq = rq[0];
        float m   = sum / (float)PIX;
        float var = (ssq - sum * sum / (float)PIX) / (float)(PIX - 1);
        meanw[nc] = m;
        rstdw[nc] = rsqrtf(var + 1e-5f);
    }
}

// ---------- 3) pointwise bias:  pw_bias = s_pool @ pb_w^T + pb_b ----------
__global__ void bias_kernel(const float* __restrict__ spool,
                            const float* __restrict__ pb_w, const float* __restrict__ pb_b,
                            float* __restrict__ biasw) {
    int g = blockIdx.x * 256 + threadIdx.x;           // g = n*512 + co
    if (g >= NB * CHN) return;
    int n = g / CHN, co = g % CHN;
    const float* sp = spool + (size_t)n * SD;
    const float* w  = pb_w + (size_t)co * SD;
    float acc = pb_b[co];
    for (int s = 0; s < SD; ++s) acc += sp[s] * w[s];
    biasw[g] = acc;
}

// ---------- 4) kernel predictor: build fused Keff (bf16, WMMA-A padded layout) ----------
__global__ void predictor_kernel(const float* __restrict__ style,
                                 const float* __restrict__ spool,
                                 const float* __restrict__ dw_w, const float* __restrict__ dw_b,
                                 const float* __restrict__ pk_w, const float* __restrict__ pk_b,
                                 unsigned short* __restrict__ keff16) {
    __shared__ float style_s[SD * 16];   // 32 KB: style[n] tile
    __shared__ float spool_s[SD];
    __shared__ float dwv[CIG * CIG * 9]; // [m][ci][tap]
    __shared__ float pwk[CIG * CIG];     // [co_local][m]

    int grp = blockIdx.x;                // 0..511 : n*64 + gg
    int n = grp / NGRP, gg = grp % NGRP;
    int tid = threadIdx.x;

    for (int i = tid; i < SD * 16; i += 256) style_s[i] = style[(size_t)n * SD * 16 + i];
    for (int i = tid; i < SD;      i += 256) spool_s[i] = spool[(size_t)n * SD + i];
    __syncthreads();

    // depthwise 3x3 kernels via 2x2 VALID conv on the 4x4 style map
    for (int o = tid; o < CIG * CIG * 9; o += 256) {
        int m = o / 72, rem = o % 72, ci = rem / 9, p = rem % 9;
        int kh = p / 3, kw = p % 3;
        int co_full = (gg * CIG + m) * CIG + ci;
        const float* wr = dw_w + (size_t)co_full * (SD * 4);
        float acc = dw_b[co_full];
        for (int s = 0; s < SD; ++s) {
            const float* st = style_s + s * 16;
            const float* w4 = wr + s * 4;
            acc += st[kh * 4 + kw]           * w4[0];
            acc += st[kh * 4 + kw + 1]       * w4[1];
            acc += st[(kh + 1) * 4 + kw]     * w4[2];
            acc += st[(kh + 1) * 4 + kw + 1] * w4[3];
        }
        dwv[o] = acc;
    }
    // pointwise kernels for this group
    for (int o = tid; o < CIG * CIG; o += 256) {
        int row = gg * 64 + o;                        // (co_local*8 + m)
        const float* w = pk_w + (size_t)row * SD;
        float acc = pk_b[row];
        for (int s = 0; s < SD; ++s) acc += spool_s[s] * w[s];
        pwk[o] = acc;
    }
    __syncthreads();

    // fuse 1x1 into 3x3, emit bf16 WMMA-A rows (rows 8..15 & K>=72 zeroed)
    unsigned short* out = keff16 + (size_t)grp * 16 * KPAD;
    for (int idx = tid; idx < 16 * KPAD; idx += 256) {
        int M = idx / KPAD, k = idx % KPAD;
        float val = 0.f;
        if (M < CIG && k < 72) {
            int ci = k / 9, p = k % 9;
            #pragma unroll
            for (int m = 0; m < CIG; ++m) val += pwk[M * CIG + m] * dwv[(m * CIG + ci) * 9 + p];
        }
        out[idx] = f2bf(val);
    }
}

// ---------- 5) fused norm + reflect-pad + grouped conv, WMMA bf16 ----------
// grid = (16 row-strips, 512 groups), 256 threads = 8 waves, 1 output row per wave
__global__ void __launch_bounds__(256)
conv_kernel(const float* __restrict__ pred,
            const float* __restrict__ meanw, const float* __restrict__ rstdw,
            const unsigned short* __restrict__ keff16,
            const float* __restrict__ biasw,
            float* __restrict__ out) {
    __shared__ unsigned short xs[XS_TOT];            // staged strip + zero pad region
    __shared__ float mloc[CIG], rloc[CIG];

    const int strip = blockIdx.x;                    // 8 output rows per strip
    const int grp   = blockIdx.y;                    // n*64 + gg
    const int n = grp / NGRP, gg = grp % NGRP;
    const int y0 = strip * 8;
    const int tid = threadIdx.x;
    const int nc0 = n * CHN + gg * CIG;

    if (tid < CIG) { mloc[tid] = meanw[nc0 + tid]; rloc[tid] = rstdw[nc0 + tid]; }
    for (int i = tid; i < XS_ZPAD; i += 256) xs[XS_MAIN + i] = 0;
    __syncthreads();

    // ---- stage normalized, reflect-padded strip into LDS as bf16 ----
    for (int idx = tid; idx < CIG * 10 * 130; idx += 256) {
        int ci  = idx / 1300;
        int rem = idx % 1300;
        int r   = rem / 130;
        int col = rem % 130;
        int gy = y0 + r - 1; if (gy < 0) gy = -gy; if (gy > HH - 1) gy = 2 * (HH - 1) - gy;
        int gx = col - 1;    if (gx < 0) gx = -gx; if (gx > WW - 1) gx = 2 * (WW - 1) - gx;
        float v = (pred[((size_t)(nc0 + ci) * HH + gy) * WW + gx] - mloc[ci]) * rloc[ci];
        xs[(ci * 10 + r) * 132 + col] = f2bf(v);
    }
    // prefetch next strip (lowers to global_prefetch_b8)
    if (strip + 1 < 16) {
        const float* pf = pred + ((size_t)nc0 * HH + (y0 + 8)) * WW;
        __builtin_prefetch(pf + tid * 16, 0, 0);
    }
    __syncthreads();

    const int lane  = tid & 31;
    const int wave  = tid >> 5;                      // == local output row rl
    const int laneM = lane & 15;
    const int hi    = lane >> 4;

    // ---- preload A = Keff (3 K-slices) in WMMA 16-bit A layout ----
    union AV { v16bf v; unsigned short u[16]; };
    AV a0, a1, a2;
    const unsigned short* ka = keff16 + (size_t)grp * 16 * KPAD + laneM * KPAD;
    #pragma unroll
    for (int j = 0; j < 16; ++j) {
        int klocal = hi * 16 + j;
        a0.u[j] = ka[klocal];
        a1.u[j] = ka[32 + klocal];
        a2.u[j] = ka[64 + klocal];
    }

    // ---- hoist all B-gather offsets: divides/preds once, zero-slot for K pad ----
    int offp[3][16];
    #pragma unroll
    for (int s = 0; s < 3; ++s) {
        #pragma unroll
        for (int j = 0; j < 16; ++j) {
            int kg = s * 32 + hi * 16 + j;           // lanes<16: K 0..15 of slice; lanes>=16: 16..31
            int o;
            if (kg < 72) {
                int ci = kg / 9, p = kg % 9;
                int kh = p / 3, kw = p % 3;
                o = (ci * 10 + kh) * 132 + kw + laneM;
            } else {
                o = XS_MAIN;                         // zeroed region
            }
            offp[s][j] = o + wave * 132;             // fold in output row
        }
    }

    float bv[CIG];
    #pragma unroll
    for (int m = 0; m < CIG; ++m) bv[m] = biasw[nc0 + m];

    const int ybase = ((y0 + wave)) * WW;

    // ---- 8 pixel tiles across the wave's row, 2 tiles per iteration ----
    for (int tx = 0; tx < WW; tx += 32) {
        v8f c0 = {}, c1 = {};
        #pragma unroll
        for (int s = 0; s < 3; ++s) {
            AV b0, b1;
            #pragma unroll
            for (int j = 0; j < 16; ++j) {
                b0.u[j] = xs[offp[s][j] + tx];
                b1.u[j] = xs[offp[s][j] + tx + 16];
            }
            const v16bf av = (s == 0) ? a0.v : (s == 1) ? a1.v : a2.v;
            c0 = __builtin_amdgcn_wmma_f32_16x16x32_bf16(false, av, false, b0.v,
                                                         (short)0, c0, false, false);
            c1 = __builtin_amdgcn_wmma_f32_16x16x32_bf16(false, av, false, b1.v,
                                                         (short)0, c1, false, false);
        }
        if (lane < 16) {
            #pragma unroll
            for (int m = 0; m < CIG; ++m) {
                float* orow = out + (size_t)(nc0 + m) * PIX + ybase;
                orow[tx + lane]      = c0[m] + bv[m];
                orow[tx + 16 + lane] = c1[m] + bv[m];
            }
        }
    }
}

extern "C" void kernel_launch(void* const* d_in, const int* in_sizes, int n_in,
                              void* d_out, int out_size, void* d_ws, size_t ws_size,
                              hipStream_t stream) {
    const float* style = (const float*)d_in[0];
    const float* pred  = (const float*)d_in[1];
    const float* dw_w  = (const float*)d_in[2];
    const float* dw_b  = (const float*)d_in[3];
    const float* pk_w  = (const float*)d_in[4];
    const float* pk_b  = (const float*)d_in[5];
    const float* pb_w  = (const float*)d_in[6];
    const float* pb_b  = (const float*)d_in[7];
    float* out = (float*)d_out;

    float* wsf = (float*)d_ws;
    float* spool = wsf + WS_SPOOL;
    float* meanw = wsf + WS_MEAN;
    float* rstdw = wsf + WS_RSTD;
    float* biasw = wsf + WS_BIAS;
    unsigned short* keff16 = (unsigned short*)((char*)d_ws + WS_KEFF_BYTES);

    spool_kernel<<<(NB * SD + 255) / 256, 256, 0, stream>>>(style, spool);
    stats_kernel<<<NB * CHN, 256, 0, stream>>>(pred, meanw, rstdw);
    bias_kernel<<<(NB * CHN + 255) / 256, 256, 0, stream>>>(spool, pb_w, pb_b, biasw);
    predictor_kernel<<<NB * NGRP, 256, 0, stream>>>(style, spool, dw_w, dw_b, pk_w, pk_b, keff16);
    conv_kernel<<<dim3(16, NB * NGRP), 256, 0, stream>>>(pred, meanw, rstdw, keff16, biasw, out);
}
